// FastAttentionBlock_26603027431580
// MI455X (gfx1250) — matrix operand order repaired and checked by reading, not compiled, not run on hardware
//
#include <hip/hip_runtime.h>
#include <hip/hip_bf16.h>
#include <hip/hip_fp16.h>

// Performer (FAVOR+) fast attention for MI455X / gfx1250, wave32 + WMMA f16.
// B=4 H=16 T=4096 DH=64 NF=256.

typedef __attribute__((ext_vector_type(16))) _Float16 v16h;
typedef __attribute__((ext_vector_type(8)))  float    v8f;

#define B_      4
#define H_      16
#define T_      4096
#define DH_     64
#define NF_     256
#define BH_     (B_ * H_)
#define TC_     128            // rows of T per block
#define NCHUNK  (T_ / TC_)     // 32 blocks per head
#define EPSV    1e-4f
#define DSCALE  0.35355339059327373f   // 64^-0.25
#define FSCALE  0.0625f                // 256^-0.5

#define CTX_N   80             // DH(64) + denom col(1) padded to 80
#define CTXF    (BH_ * NF_ * CTX_N)
#define WS_CTX_OFF 1024        // bytes: [0,256) uint max-keys, ctx after

// ---------------- WMMA fragment helpers (CDNA5 ISA layouts) ----------------
// A 16x32 f16: lanes 0-15 = rows, VGPR v<4: K=2v+8h, v>=4: K=16+2(v-4)+8h
// B 32x16 f16: lanes 0-15 = cols (lane&15), VGPR v: same K pattern as A
__device__ __forceinline__ int kpatA(int v, int h) {
  return (v < 4) ? (2 * v + 8 * h) : (16 + 2 * (v - 4) + 8 * h);
}
__device__ __forceinline__ int kpatB(int v, int h) { return kpatA(v, h); }

__device__ __forceinline__ v8f vzero8() {
  v8f z = {0.f, 0.f, 0.f, 0.f, 0.f, 0.f, 0.f, 0.f};
  return z;
}

// A(m,k) stored row-major M[m][k]
__device__ __forceinline__ v16h fragA_rowmajor(const _Float16* p, int stride,
                                               int row0, int k0, int lane) {
  int l = lane & 15, h = (lane >> 4) & 1;
  const _Float16* r = p + (row0 + l) * stride + k0;
  v16h a;
#pragma unroll
  for (int v = 0; v < 8; ++v) {
    int kk = kpatA(v, h);
    a[2 * v]     = r[kk];
    a[2 * v + 1] = r[kk + 1];
  }
  return a;
}
// A(m,k) stored k-major M[k][m]  (i.e. A = storedMatrix^T)
__device__ __forceinline__ v16h fragA_kmajor(const _Float16* p, int stride,
                                             int m0, int k0, int lane) {
  int l = lane & 15, h = (lane >> 4) & 1;
  const _Float16* c = p + k0 * stride + m0 + l;
  v16h a;
#pragma unroll
  for (int v = 0; v < 8; ++v) {
    int kk = kpatA(v, h);
    a[2 * v]     = c[kk * stride];
    a[2 * v + 1] = c[(kk + 1) * stride];
  }
  return a;
}
// B(k,col) stored col-major as M[col][k]  (e.g. proj[j][d] used as B(d,j))
__device__ __forceinline__ v16h fragB_nmajor(const _Float16* p, int stride,
                                             int k0, int col0, int lane) {
  int l = lane & 15, h = (lane >> 4) & 1;
  const _Float16* r = p + (col0 + l) * stride + k0;
  v16h b;
#pragma unroll
  for (int v = 0; v < 8; ++v) {
    int kk = kpatB(v, h);
    b[2 * v]     = r[kk];
    b[2 * v + 1] = r[kk + 1];
  }
  return b;
}
// B(k,col) stored row-major M[k][col]
__device__ __forceinline__ v16h fragB_kmajor(const _Float16* p, int stride,
                                             int k0, int col0, int lane) {
  int l = lane & 15, h = (lane >> 4) & 1;
  const _Float16* c = p + k0 * stride + col0 + l;
  v16h b;
#pragma unroll
  for (int v = 0; v < 8; ++v) {
    int kk = kpatB(v, h);
    b[2 * v]     = c[kk * stride];
    b[2 * v + 1] = c[(kk + 1) * stride];
  }
  return b;
}
__device__ __forceinline__ v8f wmma32(v16h a, v16h b, v8f c) {
  return __builtin_amdgcn_wmma_f32_16x16x32_f16(false, a, false, b, (short)0, c,
                                                false, false);
}

// order-preserving float<->uint mapping for atomicMax
__device__ __forceinline__ unsigned fkey(float f) {
  unsigned u = __float_as_uint(f);
  return (u & 0x80000000u) ? ~u : (u | 0x80000000u);
}
__device__ __forceinline__ float funkey(unsigned k) {
  unsigned u = (k & 0x80000000u) ? (k & 0x7fffffffu) : ~k;
  return __uint_as_float(u);
}

// ---------------- kernel 0: init workspace ----------------
__global__ void perf_init(unsigned* __restrict__ keys, float* __restrict__ ctx,
                          int n) {
  int i = blockIdx.x * blockDim.x + threadIdx.x;
  if (i < BH_) keys[i] = 0u;  // funkey(0) == -inf sentinel
  if (i < n) ctx[i] = 0.f;
}

// ---------------- kernel 1: k features -> context += kp^T @ [v|1] ----------
__global__ void __launch_bounds__(256) perf_k_context(
    const float* __restrict__ kglob, const float* __restrict__ vglob,
    const float* __restrict__ proj, float* __restrict__ ctxws) {
  extern __shared__ char smem[];
  _Float16* projL = (_Float16*)smem;          // [256][72]
  _Float16* ktL   = projL + 256 * 72;         // [128][72]
  _Float16* kpL   = ktL + 128 * 72;           // [128][264]
  _Float16* veL   = kpL + 128 * 264;          // [128][88]  v | ones | 0-pad
  float*    rnL   = (float*)(veL + 128 * 88); // [128]

  const int tid = threadIdx.x;
  const int wave = tid >> 5, lane = tid & 31;
  const int l = lane & 15, h = (lane >> 4) & 1;
  const int bh = blockIdx.x / NCHUNK;
  const int t0 = (blockIdx.x % NCHUNK) * TC_;

  for (int i = tid; i < NF_ * DH_; i += 256) {
    int r = i >> 6, c = i & 63;
    projL[r * 72 + c] = (_Float16)proj[i];
  }
  const float* kg = kglob + ((size_t)bh * T_ + t0) * DH_;
  for (int i = tid; i < TC_ * DH_; i += 256) {
    int r = i >> 6, c = i & 63;
    ktL[r * 72 + c] = (_Float16)(kg[i] * DSCALE);
  }
  const float* vg = vglob + ((size_t)bh * T_ + t0) * DH_;
  for (int i = tid; i < TC_ * 80; i += 256) {
    int r = i / 80, c = i - r * 80;
    float val = (c < 64) ? vg[r * 64 + c] : (c == 64 ? 1.0f : 0.0f);
    veL[r * 88 + c] = (_Float16)val;
  }
  __syncthreads();
  if (tid < TC_) {
    float s2 = 0.f;
#pragma unroll
    for (int d = 0; d < DH_; ++d) {
      float x = (float)ktL[tid * 72 + d];
      s2 += x * x;
    }
    rnL[tid] = 0.5f * s2;  // dn (tile already scaled by ds)
  }
  __syncthreads();

  // GEMM1: dp[128,256] = ktile @ proj^T ; wave owns row-tile = wave
  v8f dp[16];
  {
    v16h a0 = fragA_rowmajor(ktL, 72, wave * 16, 0, lane);
    v16h a1 = fragA_rowmajor(ktL, 72, wave * 16, 32, lane);
#pragma unroll
    for (int ct = 0; ct < 16; ++ct) {
      v8f c = vzero8();
      c = wmma32(a0, fragB_nmajor(projL, 72, 0, ct * 16, lane), c);
      c = wmma32(a1, fragB_nmajor(projL, 72, 32, ct * 16, lane), c);
      dp[ct] = c;
    }
  }
  // per-row max over NF, then kp = s*exp(dp - dn - m + eps) -> f16 LDS
#pragma unroll
  for (int r = 0; r < 8; ++r) {
    float m = dp[0][r];
#pragma unroll
    for (int ct = 1; ct < 16; ++ct) m = fmaxf(m, dp[ct][r]);
#pragma unroll
    for (int mk = 1; mk < 16; mk <<= 1) m = fmaxf(m, __shfl_xor(m, mk, 32));
    int row = wave * 16 + r + 8 * h;
    float dn = rnL[row];
#pragma unroll
    for (int ct = 0; ct < 16; ++ct) {
      float e = FSCALE * __expf(dp[ct][r] - dn - m + EPSV);
      kpL[row * 264 + ct * 16 + l] = (_Float16)e;
    }
  }
  __syncthreads();

  // GEMM2: ctx[256,80] += kp^T[256x128] @ veL[128x80]; 80 tiles / 8 waves
  float* ctx = ctxws + (size_t)bh * NF_ * CTX_N;
#pragma unroll
  for (int j = 0; j < 10; ++j) {
    int tile = wave * 10 + j;
    int mt = tile / 5, nt = tile - mt * 5;
    v8f acc = vzero8();
#pragma unroll
    for (int kk = 0; kk < 4; ++kk) {
      v16h a = fragA_kmajor(kpL, 264, mt * 16, kk * 32, lane);
      v16h b = fragB_kmajor(veL, 88, kk * 32, nt * 16, lane);
      acc = wmma32(a, b, acc);
    }
#pragma unroll
    for (int r = 0; r < 8; ++r) {
      int row = mt * 16 + r + 8 * h, col = nt * 16 + l;
      atomicAdd(&ctx[row * CTX_N + col], acc[r]);
    }
  }
}

// ---------------- kernel 2: global per-head max of dp_q ----------------
__global__ void __launch_bounds__(256) perf_q_max(
    const float* __restrict__ qglob, const float* __restrict__ proj,
    unsigned* __restrict__ keys) {
  extern __shared__ char smem[];
  _Float16* projL = (_Float16*)smem;   // [256][72]
  _Float16* qtL   = projL + 256 * 72;  // [128][72]

  const int tid = threadIdx.x;
  const int wave = tid >> 5, lane = tid & 31;
  const int bh = blockIdx.x / NCHUNK;
  const int t0 = (blockIdx.x % NCHUNK) * TC_;

  for (int i = tid; i < NF_ * DH_; i += 256) {
    int r = i >> 6, c = i & 63;
    projL[r * 72 + c] = (_Float16)proj[i];
  }
  const float* qg = qglob + ((size_t)bh * T_ + t0) * DH_;
  for (int i = tid; i < TC_ * DH_; i += 256) {
    int r = i >> 6, c = i & 63;
    qtL[r * 72 + c] = (_Float16)(qg[i] * DSCALE);
  }
  __syncthreads();

  float m = -3.0e38f;
  v16h a0 = fragA_rowmajor(qtL, 72, wave * 16, 0, lane);
  v16h a1 = fragA_rowmajor(qtL, 72, wave * 16, 32, lane);
#pragma unroll
  for (int ct = 0; ct < 16; ++ct) {
    v8f c = vzero8();
    c = wmma32(a0, fragB_nmajor(projL, 72, 0, ct * 16, lane), c);
    c = wmma32(a1, fragB_nmajor(projL, 72, 32, ct * 16, lane), c);
#pragma unroll
    for (int r = 0; r < 8; ++r) m = fmaxf(m, c[r]);
  }
#pragma unroll
  for (int mk = 1; mk < 32; mk <<= 1) m = fmaxf(m, __shfl_xor(m, mk, 32));
  if (lane == 0) atomicMax(&keys[bh], fkey(m));
}

// ---------------- kernel 3: out = (qp @ ctx) * d_inv ----------------
__global__ void __launch_bounds__(256) perf_q_out(
    const float* __restrict__ qglob, const float* __restrict__ proj,
    const float* __restrict__ ctxws, const unsigned* __restrict__ keys,
    float* __restrict__ out) {
  extern __shared__ char smem[];
  _Float16* projL = (_Float16*)smem;            // [256][72]
  _Float16* qtL   = projL + 256 * 72;           // [128][72]
  _Float16* qpL   = qtL + 128 * 72;             // [128][264]
  _Float16* ctL   = qpL + 128 * 264;            // [256][84]
  float*    rnL   = (float*)(ctL + 256 * 84);   // [128]
  float*    denL  = rnL + 128;                  // [128]

  const int tid = threadIdx.x;
  const int wave = tid >> 5, lane = tid & 31;
  const int l = lane & 15, h = (lane >> 4) & 1;
  const int bh = blockIdx.x / NCHUNK;
  const int t0 = (blockIdx.x % NCHUNK) * TC_;

  for (int i = tid; i < NF_ * DH_; i += 256) {
    int r = i >> 6, c = i & 63;
    projL[r * 72 + c] = (_Float16)proj[i];
  }
  const float* qg = qglob + ((size_t)bh * T_ + t0) * DH_;
  for (int i = tid; i < TC_ * DH_; i += 256) {
    int r = i >> 6, c = i & 63;
    qtL[r * 72 + c] = (_Float16)(qg[i] * DSCALE);
  }
  const float* cg = ctxws + (size_t)bh * NF_ * CTX_N;
  for (int i = tid; i < NF_ * CTX_N; i += 256) {
    int r = i / CTX_N, c = i - r * CTX_N;
    ctL[r * 84 + c] = (_Float16)cg[i];
  }
  const float qm = funkey(keys[bh]);
  __syncthreads();
  if (tid < TC_) {
    float s2 = 0.f;
#pragma unroll
    for (int d = 0; d < DH_; ++d) {
      float x = (float)qtL[tid * 72 + d];
      s2 += x * x;
    }
    rnL[tid] = 0.5f * s2;
  }
  __syncthreads();

  // GEMM1: dp_q, then qp = s*exp(dp - dn - qm + eps) -> f16 LDS
  {
    v16h a0 = fragA_rowmajor(qtL, 72, wave * 16, 0, lane);
    v16h a1 = fragA_rowmajor(qtL, 72, wave * 16, 32, lane);
#pragma unroll
    for (int ct = 0; ct < 16; ++ct) {
      v8f c = vzero8();
      c = wmma32(a0, fragB_nmajor(projL, 72, 0, ct * 16, lane), c);
      c = wmma32(a1, fragB_nmajor(projL, 72, 32, ct * 16, lane), c);
#pragma unroll
      for (int r = 0; r < 8; ++r) {
        int row = wave * 16 + r + 8 * h;
        float e = FSCALE * __expf(c[r] - rnL[row] - qm + EPSV);
        qpL[row * 264 + ct * 16 + l] = (_Float16)e;
      }
    }
  }
  __syncthreads();

  // GEMM2: [128,80] = qp[128,256] @ ctL[256,80]; wave owns M-tile = wave
  v8f o[5];
#pragma unroll
  for (int nt = 0; nt < 5; ++nt) o[nt] = vzero8();
#pragma unroll
  for (int kk = 0; kk < 8; ++kk) {
    v16h a = fragA_rowmajor(qpL, 264, wave * 16, kk * 32, lane);
#pragma unroll
    for (int nt = 0; nt < 5; ++nt) {
      v16h b = fragB_kmajor(ctL, 84, kk * 32, nt * 16, lane);
      o[nt] = wmma32(a, b, o[nt]);
    }
  }
  // column 64 (N-tile 4, local col 0) is the denominator qp . ksum
  if (l == 0) {
#pragma unroll
    for (int r = 0; r < 8; ++r) denL[wave * 16 + r + 8 * h] = o[4][r];
  }
  __syncthreads();

  const size_t ob = ((size_t)bh * T_ + t0) * DH_;
#pragma unroll
  for (int r = 0; r < 8; ++r) {
    int row = wave * 16 + r + 8 * h;
    float di = 1.0f / denL[row];
#pragma unroll
    for (int nt = 0; nt < 4; ++nt)
      out[ob + (size_t)row * DH_ + nt * 16 + l] = o[nt][r] * di;
  }
}

// ---------------- host launcher ----------------
extern "C" void kernel_launch(void* const* d_in, const int* in_sizes, int n_in,
                              void* d_out, int out_size, void* d_ws,
                              size_t ws_size, hipStream_t stream) {
  const float* q    = (const float*)d_in[0];
  const float* k    = (const float*)d_in[1];
  const float* v    = (const float*)d_in[2];
  const float* proj = (const float*)d_in[3];
  float* out = (float*)d_out;

  unsigned* keys = (unsigned*)d_ws;
  float* ctx = (float*)((char*)d_ws + WS_CTX_OFF);

  perf_init<<<(CTXF + 255) / 256, 256, 0, stream>>>(keys, ctx, CTXF);

  size_t lds1 = (size_t)(256 * 72 + 128 * 72 + 128 * 264 + 128 * 88) * 2 +
                128 * 4;
  perf_k_context<<<BH_ * NCHUNK, 256, lds1, stream>>>(k, v, proj, ctx);

  size_t lds2 = (size_t)(256 * 72 + 128 * 72) * 2;
  perf_q_max<<<BH_ * NCHUNK, 256, lds2, stream>>>(q, proj, keys);

  size_t lds3 = (size_t)(256 * 72 + 128 * 72 + 128 * 264 + 256 * 84) * 2 +
                128 * 4 + 128 * 4;
  perf_q_out<<<BH_ * NCHUNK, 256, lds3, stream>>>(q, proj, ctx, keys, out);
}
